// BridgeNet_FF_v2_37855841747268
// MI455X (gfx1250) — matrix operand order repaired and checked by reading, not compiled, not run on hardware
//
#include <hip/hip_runtime.h>
#include <cstdint>

typedef __attribute__((ext_vector_type(16))) _Float16 v16h;
typedef __attribute__((ext_vector_type(8)))  _Float16 v8h;
typedef __attribute__((ext_vector_type(4)))  _Float16 v4h;
typedef __attribute__((ext_vector_type(8)))  float    v8f;

#define WMMA_F16(a, b, c) \
  __builtin_amdgcn_wmma_f32_16x16x32_f16(false, (a), false, (b), (short)0, (c), false, false)

static constexpr int   NB   = 2;
static constexpr int   NN   = 16384;
static constexpr int   KKN  = 32;       // neighbors per point
static constexpr float EPSV = 1e-5f;

// ---------------------------------------------------------------------------
// Wave32 WMMA fragment layouts (f16 16x16x32):
//  A (16x32): lane l holds row (l&15); lanes 0-15 cover K 0-7 & 16-23,
//             lanes 16-31 cover K 8-15 & 24-31 -> two contiguous 8-half runs.
//  B (32x16): lane l holds col (l&15); lanes 0-15 K=0..7 & 16..23, lanes 16-31 K=8..15 & 24..31.
//  C/D (16x16 f32): lane l holds col (l&15); element j is row j + 8*(l>>4).
// B-fragments are PRE-PACKED to memory as [kstep][col][kh][16] f16 so a
// fragment is one aligned 32-byte contiguous load.
// ---------------------------------------------------------------------------

__device__ __forceinline__ v16h a_frag_f16(const _Float16* __restrict__ row,
                                           int koff, int kh) {
  v8h lo = *reinterpret_cast<const v8h*>(row + koff + kh * 8);
  v8h hi = *reinterpret_cast<const v8h*>(row + koff + 16 + kh * 8);
  v16h a;
#pragma unroll
  for (int i = 0; i < 8; ++i) { a[i] = lo[i]; a[8 + i] = hi[i]; }
  return a;
}

__device__ __forceinline__ v16h ld_bfrag(const _Float16* __restrict__ Wf, int co,
                                         int ks, int kh, int col) {
  return *reinterpret_cast<const v16h*>(Wf + (((((long)ks * co + col) << 1) + kh) << 4));
}

__device__ __forceinline__ void bn_coef(const float* bb, const float* be, const float* g,
                                        const float* m, const float* v, int c,
                                        float& scale, float& shift) {
  float s = g[c] * rsqrtf(v[c] + EPSV);
  scale = s;
  shift = (bb[c] - m[c]) * s + be[c];
}

// ---------------------------------------------------------------------------
// Prep: pack a (cin x co) f32 weight matrix into B-fragment f16 layout,
// zero-padding K up to ksteps*32.  out[e] with
// e = ((ks*co + col)*2 + kh)*16 + i ,  k = ks*32 + (i<8 ? kh*8 + i : 16 + kh*8 + i-8).
// ---------------------------------------------------------------------------
__global__ void pack_b_kernel(const float* __restrict__ W, int cin, int co, int ksteps,
                              _Float16* __restrict__ out) {
  int e = blockIdx.x * blockDim.x + threadIdx.x;
  int total = ksteps * co * 32;
  if (e >= total) return;
  int i    = e & 15;
  int kh   = (e >> 4) & 1;
  int rest = e >> 5;
  int col  = rest % co;
  int ks   = rest / co;
  int k    = ks * 32 + ((i < 8) ? (kh * 8 + i) : (16 + kh * 8 + (i - 8)));
  float w  = (k < cin) ? W[k * co + col] : 0.0f;
  out[e] = (_Float16)w;
}

// Prep: f32 -> f16 bulk convert (n multiple of 4).
__global__ void cvt_f16_kernel(const float* __restrict__ in, _Float16* __restrict__ out,
                               long n) {
  long i = ((long)blockIdx.x * blockDim.x + threadIdx.x) * 4;
  if (i >= n) return;
  float4 v = *reinterpret_cast<const float4*>(in + i);
  v4h o;
  o[0] = (_Float16)v.x; o[1] = (_Float16)v.y; o[2] = (_Float16)v.z; o[3] = (_Float16)v.w;
  *reinterpret_cast<v4h*>(out + i) = o;
}

// ---------------------------------------------------------------------------
// Kernel 1: fused geometry + pos CBN.  Rows = flattened (b,n,k) = 1,048,576.
// ---------------------------------------------------------------------------
__global__ void geo_pos_kernel(const float* __restrict__ pts,
                               const int*   __restrict__ gidx,
                               const _Float16* __restrict__ Wf,   // packed, ksteps=1, co=32
                               const float* bnb, const float* bnbe, const float* bng,
                               const float* bnm, const float* bnv,
                               _Float16* __restrict__ expanded, int ntiles) {
  const int lane = threadIdx.x & 31;
  const int wid  = threadIdx.x >> 5;
  const int tile = blockIdx.x * (blockDim.x >> 5) + wid;
  if (tile >= ntiles) return;
  const int l15 = lane & 15;
  const int kh  = lane >> 4;
  const long tb = (long)tile * 16;
  const long r  = tb + l15;                 // flattened (b,n,k)
  const int  b  = (int)(r / ((long)NN * KKN));
  const int  n  = (int)((r / KKN) % NN);
  const int  kk = (int)(r % KKN);
  const long bn = (long)b * NN + n;

  float xi0 = pts[bn * 3 + 0], xi1 = pts[bn * 3 + 1], xi2 = pts[bn * 3 + 2];
  int   gi  = gidx[bn * KKN + kk];
  long  gn  = (long)b * NN + gi;
  float gx0 = pts[gn * 3 + 0], gx1 = pts[gn * 3 + 1], gx2 = pts[gn * 3 + 2];
  float d0 = gx0 - xi0, d1 = gx1 - xi1, d2 = gx2 - xi2;
  float dist = sqrtf(d0 * d0 + d1 * d1 + d2 * d2);
  float geo[10] = {xi0, xi1, xi2, gx0, gx1, gx2, d0, d1, d2, dist};

  v16h a;
#pragma unroll
  for (int i = 0; i < 16; ++i) {
    int p = i >> 1, h = i & 1;
    int k = (p < 4) ? (kh * 8 + 2 * p + h) : (16 + kh * 8 + 2 * (p - 4) + h);
    a[i] = (k < 10) ? (_Float16)geo[k] : (_Float16)0.0f;
  }

#pragma unroll
  for (int nt = 0; nt < 2; ++nt) {
    int col = nt * 16 + l15;
    v16h bf = ld_bfrag(Wf, 32, 0, kh, col);
    v8f  c  = {};
    c = WMMA_F16(a, bf, c);
    float sc, sh;
    bn_coef(bnb, bnbe, bng, bnm, bnv, col, sc, sh);
#pragma unroll
    for (int j = 0; j < 8; ++j) {
      long row = tb + j + 8 * kh;
      float y = c[j] * sc + sh;
      y = y > 0.0f ? y : 0.0f;
      expanded[row * 32 + col] = (_Float16)y;
    }
  }
}

// ---------------------------------------------------------------------------
// Kernel 2: sub = relu(BN(f @ Wsub))   (M x ci) -> (M x 32), f16 in / f16 out
// ---------------------------------------------------------------------------
__global__ void sub_kernel(const _Float16* __restrict__ fin, int ci,
                           const _Float16* __restrict__ Wf,
                           const float* bnb, const float* bnbe, const float* bng,
                           const float* bnm, const float* bnv,
                           _Float16* __restrict__ out, int Mrows) {
  const int lane = threadIdx.x & 31;
  const int wid  = threadIdx.x >> 5;
  const int tile = blockIdx.x * (blockDim.x >> 5) + wid;
  if (tile * 16 >= Mrows) return;
  const int l15 = lane & 15, kh = lane >> 4;
  const long tb = (long)tile * 16;
  const _Float16* rp = fin + (tb + l15) * ci;

  v8f acc0 = {}, acc1 = {};
  const int nks = ci >> 5;
  for (int s = 0; s < nks; ++s) {
    v16h a  = a_frag_f16(rp, s * 32, kh);
    v16h b0 = ld_bfrag(Wf, 32, s, kh, l15);
    v16h b1 = ld_bfrag(Wf, 32, s, kh, 16 + l15);
    acc0 = WMMA_F16(a, b0, acc0);
    acc1 = WMMA_F16(a, b1, acc1);
  }
#pragma unroll
  for (int nt = 0; nt < 2; ++nt) {
    int col = nt * 16 + l15;
    float sc, sh;
    bn_coef(bnb, bnbe, bng, bnm, bnv, col, sc, sh);
    const v8f& acc = nt ? acc1 : acc0;
#pragma unroll
    for (int j = 0; j < 8; ++j) {
      float y = acc[j] * sc + sh;
      y = y > 0.0f ? y : 0.0f;
      out[(tb + j + 8 * kh) * 32 + col] = (_Float16)y;
    }
  }
}

// ---------------------------------------------------------------------------
// Kernel 3: per-point gcm GEMM (32x64 @ 64x32) + BN/ReLU + channel softmax
// attention + K-maxpool.  One wave per point; 8 WMMAs.
// ---------------------------------------------------------------------------
__global__ void gcm_kernel(const _Float16* __restrict__ expanded,
                           const _Float16* __restrict__ sub,
                           const int*      __restrict__ gidx,
                           const _Float16* __restrict__ Wf,  // packed, ksteps=2, co=32
                           const float* bnb, const float* bnbe, const float* bng,
                           const float* bnm, const float* bnv,
                           const float* __restrict__ attW, const float* __restrict__ attb,
                           _Float16* __restrict__ pooled, int Mrows) {
  const int lane = threadIdx.x & 31;
  const int wid  = threadIdx.x >> 5;
  const long bn  = (long)blockIdx.x * (blockDim.x >> 5) + wid;
  if (bn >= Mrows) return;
  const int l15 = lane & 15, kh = lane >> 4;
  const int b   = (int)(bn / NN);

  v16h b00 = ld_bfrag(Wf, 32, 0, kh, l15);
  v16h b01 = ld_bfrag(Wf, 32, 0, kh, 16 + l15);
  v16h b10 = ld_bfrag(Wf, 32, 1, kh, l15);
  v16h b11 = ld_bfrag(Wf, 32, 1, kh, 16 + l15);

  v8f acc00 = {}, acc01 = {}, acc10 = {}, acc11 = {};
#pragma unroll
  for (int mt = 0; mt < 2; ++mt) {
    int  r  = mt * 16 + l15;                                      // neighbor index
    v16h a0 = a_frag_f16(expanded + (bn * 32 + r) * 32, 0, kh);   // expanded_f row
    int  gi = gidx[bn * 32 + r];
    v16h a1 = a_frag_f16(sub + ((long)b * NN + gi) * 32, 0, kh);  // gathered sub row
    if (mt == 0) {
      acc00 = WMMA_F16(a0, b00, acc00); acc00 = WMMA_F16(a1, b10, acc00);
      acc01 = WMMA_F16(a0, b01, acc01); acc01 = WMMA_F16(a1, b11, acc01);
    } else {
      acc10 = WMMA_F16(a0, b00, acc10); acc10 = WMMA_F16(a1, b10, acc10);
      acc11 = WMMA_F16(a0, b01, acc11); acc11 = WMMA_F16(a1, b11, acc11);
    }
  }

  float sc0, sh0, sc1, sh1;
  bn_coef(bnb, bnbe, bng, bnm, bnv, l15,      sc0, sh0);
  bn_coef(bnb, bnbe, bng, bnm, bnv, 16 + l15, sc1, sh1);

  float cf0[2][8], cf1[2][8];   // [mtile][j] for cols l15 and 16+l15
#pragma unroll
  for (int j = 0; j < 8; ++j) {
    float y;
    y = acc00[j] * sc0 + sh0; cf0[0][j] = y > 0.0f ? y : 0.0f;
    y = acc10[j] * sc0 + sh0; cf0[1][j] = y > 0.0f ? y : 0.0f;
    y = acc01[j] * sc1 + sh1; cf1[0][j] = y > 0.0f ? y : 0.0f;
    y = acc11[j] * sc1 + sh1; cf1[1][j] = y > 0.0f ? y : 0.0f;
  }

  // attention logits (sum_k cf*attW) and K-maxpool over this lane's 16 rows
  float s0 = 0.0f, s1 = 0.0f, p0 = 0.0f, p1 = 0.0f;
#pragma unroll
  for (int mt = 0; mt < 2; ++mt) {
#pragma unroll
    for (int j = 0; j < 8; ++j) {
      float aw = attW[mt * 16 + j + 8 * kh];
      s0 += cf0[mt][j] * aw;  s1 += cf1[mt][j] * aw;
      p0 = fmaxf(p0, cf0[mt][j]);  p1 = fmaxf(p1, cf1[mt][j]);
    }
  }
  // combine with partner half (rows covered by lane^16)
  s0 += __shfl_xor(s0, 16, 32);
  s1 += __shfl_xor(s1, 16, 32);
  p0 = fmaxf(p0, __shfl_xor(p0, 16, 32));
  p1 = fmaxf(p1, __shfl_xor(p1, 16, 32));
  float ab = attb[0];
  s0 += ab; s1 += ab;

  // softmax over the 32 channels (distributed as (s0,s1) across 16-lane group)
  float gm = fmaxf(s0, s1);
#pragma unroll
  for (int msk = 1; msk < 16; msk <<= 1) gm = fmaxf(gm, __shfl_xor(gm, msk, 32));
  float e0 = __expf(s0 - gm), e1 = __expf(s1 - gm);
  float den = e0 + e1;
#pragma unroll
  for (int msk = 1; msk < 16; msk <<= 1) den += __shfl_xor(den, msk, 32);

  float o0 = p0 * (1.0f + e0 / den);
  float o1 = p1 * (1.0f + e1 / den);
  if (kh == 0) {
    pooled[bn * 32 + l15]      = (_Float16)o0;
    pooled[bn * 32 + 16 + l15] = (_Float16)o1;
  }
}

// ---------------------------------------------------------------------------
// Kernel 4: f_out = relu(BN(f @ Wres)) + relu(BN(pooled @ Wup)), co = 128.
// Writes f16 (intermediate, block 0) or f32 (final, block 1).
// ---------------------------------------------------------------------------
__global__ void out_kernel(const _Float16* __restrict__ fin, int ci,
                           const _Float16* __restrict__ pooled,
                           const _Float16* __restrict__ WresF,
                           const float* rb, const float* rbe, const float* rg,
                           const float* rm, const float* rv,
                           const _Float16* __restrict__ WupF,
                           const float* ub, const float* ube, const float* ug,
                           const float* um, const float* uv,
                           float* __restrict__ fout32, _Float16* __restrict__ fout16,
                           int Mrows) {
  const int lane = threadIdx.x & 31;
  const int wid  = threadIdx.x >> 5;
  const int task = blockIdx.x * (blockDim.x >> 5) + wid;
  const int mtile = task >> 2;
  const int cb    = (task & 3) * 32;
  if (mtile * 16 >= Mrows) return;
  const int l15 = lane & 15, kh = lane >> 4;
  const long tb = (long)mtile * 16;
  const _Float16* rp = fin + (tb + l15) * ci;

  // residual branch
  v8f acc0 = {}, acc1 = {};
  const int nks = ci >> 5;
  for (int s = 0; s < nks; ++s) {
    v16h a  = a_frag_f16(rp, s * 32, kh);
    v16h b0 = ld_bfrag(WresF, 128, s, kh, cb + l15);
    v16h b1 = ld_bfrag(WresF, 128, s, kh, cb + 16 + l15);
    acc0 = WMMA_F16(a, b0, acc0);
    acc1 = WMMA_F16(a, b1, acc1);
  }
  float outv0[8], outv1[8];
  {
    float sc, sh;
    bn_coef(rb, rbe, rg, rm, rv, cb + l15, sc, sh);
#pragma unroll
    for (int j = 0; j < 8; ++j) { float y = acc0[j] * sc + sh; outv0[j] = y > 0.0f ? y : 0.0f; }
    bn_coef(rb, rbe, rg, rm, rv, cb + 16 + l15, sc, sh);
#pragma unroll
    for (int j = 0; j < 8; ++j) { float y = acc1[j] * sc + sh; outv1[j] = y > 0.0f ? y : 0.0f; }
  }

  // up branch (K = 32, single step)
  v8f u0 = {}, u1 = {};
  {
    v16h a  = a_frag_f16(pooled + (tb + l15) * 32, 0, kh);
    v16h b0 = ld_bfrag(WupF, 128, 0, kh, cb + l15);
    v16h b1 = ld_bfrag(WupF, 128, 0, kh, cb + 16 + l15);
    u0 = WMMA_F16(a, b0, u0);
    u1 = WMMA_F16(a, b1, u1);
  }
  {
    float sc, sh;
    bn_coef(ub, ube, ug, um, uv, cb + l15, sc, sh);
#pragma unroll
    for (int j = 0; j < 8; ++j) { float y = u0[j] * sc + sh; outv0[j] += y > 0.0f ? y : 0.0f; }
    bn_coef(ub, ube, ug, um, uv, cb + 16 + l15, sc, sh);
#pragma unroll
    for (int j = 0; j < 8; ++j) { float y = u1[j] * sc + sh; outv1[j] += y > 0.0f ? y : 0.0f; }
  }

  if (fout32 != nullptr) {
#pragma unroll
    for (int j = 0; j < 8; ++j) {
      long row = tb + j + 8 * kh;
      fout32[row * 128 + cb + l15]      = outv0[j];
      fout32[row * 128 + cb + 16 + l15] = outv1[j];
    }
  } else {
#pragma unroll
    for (int j = 0; j < 8; ++j) {
      long row = tb + j + 8 * kh;
      fout16[row * 128 + cb + l15]      = (_Float16)outv0[j];
      fout16[row * 128 + cb + 16 + l15] = (_Float16)outv1[j];
    }
  }
}

// ---------------------------------------------------------------------------
// Launcher
// ---------------------------------------------------------------------------
extern "C" void kernel_launch(void* const* d_in, const int* in_sizes, int n_in,
                              void* d_out, int out_size, void* d_ws, size_t ws_size,
                              hipStream_t stream) {
  (void)out_size; (void)ws_size;
  // ---- detect top-level ordering ----
  int ipts, ifeat, igidx, pbase;
  if (in_sizes[0] == NB * NN * 3) {            // insertion order: points first
    ipts = 0; ifeat = 1; igidx = 2; pbase = 3;
  } else {                                     // jax sorted order: points last
    ifeat = 0; igidx = 1; pbase = 2; ipts = n_in - 1;
  }
  // ---- detect params flattening scheme ----
  const bool sorted_params = (in_sizes[pbase] == 32);  // attW(32) first if sorted
  int posBase, blkBase[2], gcmOff, resOff, subOff, upOff, attWOff, attbOff;
  int cbnMap[6]; // leaf order: 0=W,1=b,2=be,3=g,4=m,5=v
  if (sorted_params) {
    blkBase[0] = 0; blkBase[1] = 26; posBase = 52;
    attWOff = 0; attbOff = 1; gcmOff = 2; resOff = 8; subOff = 14; upOff = 20;
    int mp[6] = {0, 1, 2, 3, 4, 5};
    for (int i = 0; i < 6; ++i) cbnMap[i] = mp[i];
  } else {  // insertion order: pos first; cbn order W,b,g,be,m,v; blk: sub,gcm,attW,attb,up,res
    posBase = 0; blkBase[0] = 6; blkBase[1] = 32;
    subOff = 0; gcmOff = 6; attWOff = 12; attbOff = 13; upOff = 14; resOff = 20;
    int mp[6] = {0, 1, 3, 2, 4, 5};
    for (int i = 0; i < 6; ++i) cbnMap[i] = mp[i];
  }
  auto P = [&](int off) -> const float* { return (const float*)d_in[pbase + off]; };

  const float* posP[6];
  for (int l = 0; l < 6; ++l) posP[l] = P(posBase + cbnMap[l]);
  const float *gcmP[2][6], *resP[2][6], *subP[2][6], *upP[2][6], *attWP[2], *attbP[2];
  for (int i = 0; i < 2; ++i) {
    attWP[i] = P(blkBase[i] + attWOff);
    attbP[i] = P(blkBase[i] + attbOff);
    for (int l = 0; l < 6; ++l) {
      gcmP[i][l] = P(blkBase[i] + gcmOff + cbnMap[l]);
      resP[i][l] = P(blkBase[i] + resOff + cbnMap[l]);
      subP[i][l] = P(blkBase[i] + subOff + cbnMap[l]);
      upP[i][l]  = P(blkBase[i] + upOff  + cbnMap[l]);
    }
  }

  const float* pts  = (const float*)d_in[ipts];
  const float* feat = (const float*)d_in[ifeat];
  const int*   gidx = (const int*)  d_in[igidx];

  // ---- workspace layout (bytes) ----
  char* ws = (char*)d_ws;
  _Float16* expanded = (_Float16*)(ws);                       // 64 MB
  _Float16* subws    = (_Float16*)(ws + 67108864ull);         //  2 MB
  _Float16* pooledws = (_Float16*)(ws + 69206016ull);         //  2 MB
  _Float16* f1h      = (_Float16*)(ws + 71303168ull);         //  8 MB (block0 out, f16)
  _Float16* featH    = (_Float16*)(ws + 79691776ull);         //  4 MB (features, f16)
  _Float16* wpack    = (_Float16*)(ws + 83886080ull);         //  < 128 KB packed weights

  // packed-weight sub-offsets (in halves): size = ksteps*co*32
  _Float16* posF    = wpack + 0;        // 1*32*32   = 1024
  _Float16* subF[2] = { wpack + 1024,   // 2*32*32   = 2048
                        wpack + 3072 }; // 4*32*32   = 4096
  _Float16* gcmF[2] = { wpack + 7168,   // 2*32*32   = 2048
                        wpack + 9216 }; // 2*32*32   = 2048
  _Float16* resF[2] = { wpack + 11264,  // 2*128*32  = 8192
                        wpack + 19456 };// 4*128*32  = 16384
  _Float16* upF[2]  = { wpack + 35840,  // 1*128*32  = 4096
                        wpack + 39936 };// 1*128*32  = 4096

  const int ciArr[2] = {64, 128};
  const int M        = NB * NN;              // 32768 points
  const int geoTil   = NB * NN * KKN / 16;   // 65536 tiles

  // ---- prep: pack weights + convert features ----
  auto packLaunch = [&](const float* W, int cin, int co, int ks, _Float16* out) {
    int total = ks * co * 32;
    pack_b_kernel<<<(total + 255) / 256, 256, 0, stream>>>(W, cin, co, ks, out);
  };
  packLaunch(posP[0], 10, 32, 1, posF);
  for (int i = 0; i < 2; ++i) {
    packLaunch(subP[i][0], ciArr[i], 32, ciArr[i] / 32, subF[i]);
    packLaunch(gcmP[i][0], 64, 32, 2, gcmF[i]);
    packLaunch(resP[i][0], ciArr[i], 128, ciArr[i] / 32, resF[i]);
    packLaunch(upP[i][0], 32, 128, 1, upF[i]);
  }
  {
    long n = (long)M * 64;
    cvt_f16_kernel<<<(int)((n / 4 + 255) / 256), 256, 0, stream>>>(feat, featH, n);
  }

  // ---- stage 1: geometry + pos ----
  geo_pos_kernel<<<geoTil / 8, 256, 0, stream>>>(
      pts, gidx, posF, posP[1], posP[2], posP[3], posP[4], posP[5], expanded, geoTil);

  // ---- stage 2: two blocks ----
  const _Float16* finArr[2] = {featH, f1h};
  for (int i = 0; i < 2; ++i) {
    const int ci = ciArr[i];
    sub_kernel<<<(M / 16) / 8, 256, 0, stream>>>(
        finArr[i], ci, subF[i], subP[i][1], subP[i][2], subP[i][3], subP[i][4], subP[i][5],
        subws, M);
    gcm_kernel<<<M / 8, 256, 0, stream>>>(
        expanded, subws, gidx, gcmF[i], gcmP[i][1], gcmP[i][2], gcmP[i][3], gcmP[i][4],
        gcmP[i][5], attWP[i], attbP[i], pooledws, M);
    out_kernel<<<(M / 16) * 4 / 8, 256, 0, stream>>>(
        finArr[i], ci, pooledws,
        resF[i], resP[i][1], resP[i][2], resP[i][3], resP[i][4], resP[i][5],
        upF[i],  upP[i][1],  upP[i][2],  upP[i][3],  upP[i][4],  upP[i][5],
        (i == 1) ? (float*)d_out : nullptr, (i == 0) ? f1h : nullptr, M);
  }
}
